// attention_85358180041208
// MI455X (gfx1250) — compile-verified, matrix operand order checked by reading, not run on hardware
//
#include <hip/hip_runtime.h>
#include <hip/hip_bf16.h>
#include <stdint.h>

// Problem constants (from reference)
#define D_   1024
#define H_   16
#define HD_  64
#define F_   32
#define B_   2
#define C_   2048
#define BC_  (B_ * C_)

typedef __attribute__((ext_vector_type(16))) __bf16 v16bf;
typedef __attribute__((ext_vector_type(8)))  float  v8f;

#define GLOBAL_AS __attribute__((address_space(1)))
#define LDS_AS    __attribute__((address_space(3)))

// --- CDNA5 feature detection (fallbacks keep the file compiling anywhere) ---
#if defined(__has_builtin)
#if __has_builtin(__builtin_amdgcn_global_load_async_to_lds_b128)
#define HAVE_ASYNC 1
#endif
#if __has_builtin(__builtin_amdgcn_tensor_load_to_lds)
#define HAVE_TDM 1
#endif
#endif

typedef unsigned int u32x4 __attribute__((ext_vector_type(4)));
typedef int          i32x4 __attribute__((ext_vector_type(4)));
typedef int          i32x8 __attribute__((ext_vector_type(8)));

static __device__ __forceinline__ unsigned lds_off_u32(void* p) {
  return (unsigned)(unsigned long long)(LDS_AS void*)p;
}

static __device__ __forceinline__ void async_ld_b128(const void* g, void* l) {
#if defined(HAVE_ASYNC)
  __builtin_amdgcn_global_load_async_to_lds_b128(
      (GLOBAL_AS i32x4*)(size_t)g, (LDS_AS i32x4*)(LDS_AS void*)l, 0, 0);
#else
  (void)g; (void)l;
#endif
}

template <int N> static __device__ __forceinline__ void wait_async() {
#if defined(HAVE_ASYNC)
#if __has_builtin(__builtin_amdgcn_s_wait_asynccnt)
  __builtin_amdgcn_s_wait_asynccnt(N);
#else
  asm volatile("s_wait_asynccnt %0" ::"n"(N) : "memory");
#endif
#endif
}

template <int N> static __device__ __forceinline__ void wait_tensor() {
#if defined(HAVE_TDM)
#if __has_builtin(__builtin_amdgcn_s_wait_tensorcnt)
  __builtin_amdgcn_s_wait_tensorcnt(N);
#else
  asm volatile("s_wait_tensorcnt %0" ::"n"(N) : "memory");
#endif
#endif
}

#if defined(HAVE_TDM)
// 2-D tile load via the Tensor Data Mover. Dims/strides in bf16 elements
// (data_size=1 -> 2 bytes); addresses in bytes. D# per CDNA5 ISA.
static __device__ __forceinline__ void tdm_load_2d(
    const void* gptr, unsigned lds_byte_off,
    unsigned tile_d0, unsigned tile_d1,
    unsigned tensor_d0, unsigned tensor_d1,
    unsigned long long stride0_elems) {
  const unsigned long long ga = (unsigned long long)(size_t)gptr;
  u32x4 g0;
  g0.x = 1u;                                        // count=1, user descriptor
  g0.y = lds_byte_off;                              // lds_addr
  g0.z = (unsigned)(ga & 0xFFFFFFFFull);            // global_addr[31:0]
  g0.w = (unsigned)((ga >> 32) & 0x1FFFFFFull)      // global_addr[56:32]
         | (2u << 30);                              // type = 2 ("image")
  i32x8 g1;
  g1[0] = (int)(1u << 16);                          // wg_mask=0, data_size=1 (2B)
  g1[1] = (int)((tensor_d0 & 0xFFFFu) << 16);       // tensor_dim0[15:0]
  g1[2] = (int)(((tensor_d0 >> 16) & 0xFFFFu) | ((tensor_d1 & 0xFFFFu) << 16));
  g1[3] = (int)(((tensor_d1 >> 16) & 0xFFFFu) | ((tile_d0 & 0xFFFFu) << 16));
  g1[4] = (int)(tile_d1 & 0xFFFFu);                 // tile_dim1; tile_dim2=0
  g1[5] = (int)(unsigned)(stride0_elems & 0xFFFFFFFFull);
  g1[6] = (int)(unsigned)((stride0_elems >> 32) & 0xFFFFull);
  g1[7] = 0;
  const i32x4 z4 = {0, 0, 0, 0};
#if __clang_major__ >= 23
  const i32x8 z8 = {0, 0, 0, 0, 0, 0, 0, 0};
  __builtin_amdgcn_tensor_load_to_lds(g0, g1, z4, z4, z8, 0);
#else
  __builtin_amdgcn_tensor_load_to_lds(g0, g1, z4, z4, 0);
#endif
}
#endif

static __device__ __forceinline__ unsigned short f2bf(float f) {
  union { float f; unsigned u; } cv; cv.f = f;
  unsigned u = cv.u;
  return (unsigned short)((u + 0x7FFFu + ((u >> 16) & 1u)) >> 16); // RNE
}

static __device__ __forceinline__ v8f wmma_bf16(v16bf a, v16bf b, v8f c) {
  return __builtin_amdgcn_wmma_f32_16x16x32_bf16(false, a, false, b, (short)0, c,
                                                 false, false);
}

static __device__ __forceinline__ float wave_sum(float v) {
  #pragma unroll
  for (int o = 16; o > 0; o >>= 1) v += __shfl_xor(v, o, 32);
  return v;
}

// ---------------------------------------------------------------------------
// Kernel 1: per-row LN stats (shared by LN_q / LN_kv) + radius head.
// ---------------------------------------------------------------------------
__global__ __launch_bounds__(256)
void k_prep(const float* __restrict__ x,
            const float* __restrict__ gq,  const float* __restrict__ bq,
            const float* __restrict__ gkv, const float* __restrict__ bkv,
            const float* __restrict__ Wr,  const float* __restrict__ br,
            unsigned short* __restrict__ lnq,
            unsigned short* __restrict__ lnkv,
            float* __restrict__ radius) {
  const int row = blockIdx.x;
  const int t   = threadIdx.x;
  const float4 xv = ((const float4*)(x + (size_t)row * D_))[t];
  const float4 wr = ((const float4*)Wr)[t];

  float s  = xv.x + xv.y + xv.z + xv.w;
  float s2 = xv.x * xv.x + xv.y * xv.y + xv.z * xv.z + xv.w * xv.w;
  float sr = xv.x * wr.x + xv.y * wr.y + xv.z * wr.z + xv.w * wr.w;

  __shared__ float red[3][8];
  const int lane = t & 31, wid = t >> 5;
  s = wave_sum(s); s2 = wave_sum(s2); sr = wave_sum(sr);
  if (lane == 0) { red[0][wid] = s; red[1][wid] = s2; red[2][wid] = sr; }
  __syncthreads();
  float S = 0.f, S2 = 0.f, SR = 0.f;
  #pragma unroll
  for (int i = 0; i < 8; ++i) { S += red[0][i]; S2 += red[1][i]; SR += red[2][i]; }

  const float mean = S * (1.0f / (float)D_);
  const float var  = S2 * (1.0f / (float)D_) - mean * mean;
  const float rstd = rsqrtf(var + 1e-5f);

  const float xe[4] = { xv.x, xv.y, xv.z, xv.w };
  #pragma unroll
  for (int i = 0; i < 4; ++i) {
    const int c = t * 4 + i;
    const float xn = (xe[i] - mean) * rstd;
    lnq [(size_t)row * D_ + c] = f2bf(xn * gq[c]  + bq[c]);
    lnkv[(size_t)row * D_ + c] = f2bf(xn * gkv[c] + bkv[c]);
  }
  if (t == 0) {
    const float ps = (SR + br[0]) * 0.01f;
    radius[row] = fminf(2.0f, fmaxf(0.5f, ps));
  }
}

// ---------------------------------------------------------------------------
// Kernel 2: transpose-convert weights f32 [K,N] -> bf16 [N,K].
// ---------------------------------------------------------------------------
__global__ void k_w2bf_t(const float* __restrict__ in,
                         unsigned short* __restrict__ out, int K, int N) {
  const int idx = blockIdx.x * blockDim.x + threadIdx.x;
  if (idx < N * K) {
    const int n = idx / K, k = idx - n * K;
    out[idx] = f2bf(in[(size_t)k * N + n]);
  }
}

// ---------------------------------------------------------------------------
// Kernel 3: bf16 WMMA GEMM, Out[MxN](f32) = A[MxK] @ Wt[NxK]^T + bias.
// 128x128 tile, K-step 64, double-buffered async global->LDS staging.
// 16 WMMAs per barrier pair.
// ---------------------------------------------------------------------------
static __device__ __forceinline__ void gemm_stage(
    const unsigned short* A, const unsigned short* Wt, int K,
    int m0, int n0, int k0, unsigned short* aBuf, unsigned short* wBuf, int t) {
  const int r  = t >> 1;
  const int hh = (t & 1) * 32;
  const unsigned short* ag = A  + (size_t)(m0 + r) * K + k0 + hh;
  const unsigned short* wg = Wt + (size_t)(n0 + r) * K + k0 + hh;
  unsigned short* al = aBuf + r * 64 + hh;
  unsigned short* wl = wBuf + r * 64 + hh;
#if defined(HAVE_ASYNC)
  #pragma unroll
  for (int i = 0; i < 4; ++i) {
    async_ld_b128(ag + i * 8, al + i * 8);
    async_ld_b128(wg + i * 8, wl + i * 8);
  }
#else
  #pragma unroll
  for (int i = 0; i < 4; ++i) {
    ((uint4*)al)[i] = ((const uint4*)ag)[i];
    ((uint4*)wl)[i] = ((const uint4*)wg)[i];
  }
#endif
}

__global__ __launch_bounds__(256)
void k_gemm(const unsigned short* __restrict__ A,
            const unsigned short* __restrict__ Wt,
            const float* __restrict__ bias,
            float* __restrict__ Out, int M, int N, int K) {
  const int n0 = blockIdx.x * 128;
  const int m0 = blockIdx.y * 128;
  const int t    = threadIdx.x;
  const int lane = t & 31;
  const int wid  = t >> 5;
  const int wrow = wid >> 2;   // 0..1 -> 64 rows
  const int wcol = wid & 3;    // 0..3 -> 32 cols

  __shared__ __align__(32) unsigned short As[2][128 * 64];  // 16 KB each
  __shared__ __align__(32) unsigned short Ws[2][128 * 64];  // 16 KB each

  v8f acc[4][2];
  #pragma unroll
  for (int i = 0; i < 4; ++i)
    #pragma unroll
    for (int j = 0; j < 2; ++j)
      acc[i][j] = (v8f){0.f,0.f,0.f,0.f,0.f,0.f,0.f,0.f};

  int cur = 0;
  gemm_stage(A, Wt, K, m0, n0, 0, &As[0][0], &Ws[0][0], t);

  for (int k0 = 0; k0 < K; k0 += 64) {
    const bool more = (k0 + 64) < K;
    if (more) {
      gemm_stage(A, Wt, K, m0, n0, k0 + 64, &As[cur ^ 1][0], &Ws[cur ^ 1][0], t);
      wait_async<8>();
    } else {
      wait_async<0>();
    }
    __syncthreads();

    const int kh = (lane >> 4) * 16;
    #pragma unroll
    for (int kk = 0; kk < 2; ++kk) {
      const int kb = kk * 32 + kh;
      v16bf af[4];
      #pragma unroll
      for (int mt = 0; mt < 4; ++mt) {
        const int r = wrow * 64 + mt * 16 + (lane & 15);
        af[mt] = *(const v16bf*)(&As[cur][r * 64 + kb]);
      }
      v16bf bf2[2];
      #pragma unroll
      for (int nt = 0; nt < 2; ++nt) {
        const int c = wcol * 32 + nt * 16 + (lane & 15);
        bf2[nt] = *(const v16bf*)(&Ws[cur][c * 64 + kb]);
      }
      #pragma unroll
      for (int mt = 0; mt < 4; ++mt)
        #pragma unroll
        for (int nt = 0; nt < 2; ++nt)
          acc[mt][nt] = wmma_bf16(af[mt], bf2[nt], acc[mt][nt]);
    }
    __syncthreads();
    cur ^= 1;
  }

  #pragma unroll
  for (int mt = 0; mt < 4; ++mt) {
    #pragma unroll
    for (int nt = 0; nt < 2; ++nt) {
      const int col = n0 + wcol * 32 + nt * 16 + (lane & 15);
      const float bv = bias[col];
      const int mbase = m0 + wrow * 64 + mt * 16 + ((lane >> 4) ? 8 : 0);
      #pragma unroll
      for (int e = 0; e < 8; ++e)
        Out[(size_t)(mbase + e) * N + col] = acc[mt][nt][e] + bv;
    }
  }
}

// ---------------------------------------------------------------------------
// Kernel 4: RoPE-polar + per-head LN; qn,kn in [B,H,C,HD]; V in [B,H,HD,C].
// ---------------------------------------------------------------------------
__global__ __launch_bounds__(32)
void k_rope_ln(const float* __restrict__ q, const float* __restrict__ kv,
               const float* __restrict__ radius,
               const float* __restrict__ lg, const float* __restrict__ lb,
               unsigned short* __restrict__ qn,
               unsigned short* __restrict__ kn,
               unsigned short* __restrict__ vbT) {
  const int bid = blockIdx.x;
  const int h  = bid & (H_ - 1);
  const int bc = bid >> 4;
  const int c  = bc & (C_ - 1);
  const int b  = bc >> 11;
  const int f  = threadIdx.x;

  const float freq = 0.2f * (exp2f((float)f * (4.392317422778761f / 31.0f)) - 1.0f);
  float sn, cs;
  __sincosf((float)c * freq, &sn, &cs);
  const float r = radius[bc];
  const float g0 = lg[2 * f], g1 = lg[2 * f + 1];
  const float bb0 = lb[2 * f], bb1 = lb[2 * f + 1];
  const size_t hoff = ((size_t)(b * H_ + h) * C_ + c) * HD_ + 2 * f;

  { // q
    const size_t off = (size_t)bc * D_ + h * HD_ + 2 * f;
    const float xr = q[off], xi = q[off + 1];
    const float orr = r * (xr * cs - xi * sn);
    const float oii = r * (xr * sn + xi * cs);
    const float mean = wave_sum(orr + oii) * (1.0f / (float)HD_);
    const float d0 = orr - mean, d1 = oii - mean;
    const float var = wave_sum(d0 * d0 + d1 * d1) * (1.0f / (float)HD_);
    const float rstd = rsqrtf(var + 1e-5f);
    qn[hoff]     = f2bf(d0 * rstd * g0 + bb0);
    qn[hoff + 1] = f2bf(d1 * rstd * g1 + bb1);
  }
  { // k
    const size_t off = (size_t)bc * (2 * D_) + h * HD_ + 2 * f;
    const float xr = kv[off], xi = kv[off + 1];
    const float orr = r * (xr * cs - xi * sn);
    const float oii = r * (xr * sn + xi * cs);
    const float mean = wave_sum(orr + oii) * (1.0f / (float)HD_);
    const float d0 = orr - mean, d1 = oii - mean;
    const float var = wave_sum(d0 * d0 + d1 * d1) * (1.0f / (float)HD_);
    const float rstd = rsqrtf(var + 1e-5f);
    kn[hoff]     = f2bf(d0 * rstd * g0 + bb0);
    kn[hoff + 1] = f2bf(d1 * rstd * g1 + bb1);
  }
  { // v -> bf16, transposed [B,H,HD,C]
    const size_t off = (size_t)bc * (2 * D_) + D_ + h * HD_ + 2 * f;
    const size_t vo = ((size_t)(b * H_ + h) * HD_ + 2 * f) * C_ + c;
    vbT[vo]      = f2bf(kv[off]);
    vbT[vo + C_] = f2bf(kv[off + 1]);
  }
}

// ---------------------------------------------------------------------------
// Kernel 5: flash attention; K/V chunk staging via TDM (preferred) or
// async-to-LDS, double buffered. 16 WMMAs per 64-key chunk per wave.
// ---------------------------------------------------------------------------
static __device__ __forceinline__ void attn_stage(
    const unsigned short* Kp, const unsigned short* Vt, int kc,
    unsigned short* ksBuf, unsigned short* vsBuf, int t, int wid) {
#if defined(HAVE_TDM)
  if (wid == 0) {
    // K chunk: rows = keys (stride HD_), tile 64x64, contiguous.
    tdm_load_2d(Kp + (size_t)kc * HD_, lds_off_u32(ksBuf),
                /*tile_d0=*/HD_, /*tile_d1=*/64,
                /*tensor_d0=*/HD_, /*tensor_d1=*/C_, /*stride0=*/HD_);
    // V chunk: rows = d (stride C_), cols = keys kc..kc+63.
    tdm_load_2d(Vt + kc, lds_off_u32(vsBuf),
                /*tile_d0=*/64, /*tile_d1=*/HD_,
                /*tensor_d0=*/C_, /*tensor_d1=*/HD_, /*stride0=*/C_);
  }
#else
  const int r = t >> 1, off = (t & 1) * 32;
  const unsigned short* kg = Kp + (size_t)(kc + r) * HD_ + off;
  const unsigned short* vg = Vt + (size_t)r * C_ + kc + off;
  unsigned short* kl = ksBuf + r * 64 + off;
  unsigned short* vl = vsBuf + r * 64 + off;
#if defined(HAVE_ASYNC)
  #pragma unroll
  for (int i = 0; i < 4; ++i) {
    async_ld_b128(kg + i * 8, kl + i * 8);
    async_ld_b128(vg + i * 8, vl + i * 8);
  }
#else
  #pragma unroll
  for (int i = 0; i < 4; ++i) {
    ((uint4*)kl)[i] = ((const uint4*)kg)[i];
    ((uint4*)vl)[i] = ((const uint4*)vg)[i];
  }
#endif
  (void)wid;
#endif
}

__global__ __launch_bounds__(128)
void k_attn(const unsigned short* __restrict__ qn,
            const unsigned short* __restrict__ kn,
            const unsigned short* __restrict__ vbT,
            unsigned short* __restrict__ obf) {
  const int bh   = blockIdx.x >> 5;
  const int qblk = blockIdx.x & 31;
  const int t    = threadIdx.x;
  const int lane = t & 31;
  const int wid  = t >> 5;
  const int qbase = qblk * 64 + wid * 16;

  const unsigned short* Qp = qn  + (size_t)bh * C_ * HD_;
  const unsigned short* Kp = kn  + (size_t)bh * C_ * HD_;
  const unsigned short* Vt = vbT + (size_t)bh * HD_ * C_;

  __shared__ __align__(32) unsigned short Ks [2][64 * 64];
  __shared__ __align__(32) unsigned short Vst[2][64 * 64];
  __shared__ __align__(32) float          Sl [4][16 * 64];
  __shared__ __align__(32) unsigned short Pl [4][16 * 64];
  __shared__ float alpha_s[4][16];
  __shared__ float linv_s [4][16];

  const int kh = (lane >> 4) * 16;
  v16bf aq[2];
  {
    const int row = qbase + (lane & 15);
    aq[0] = *(const v16bf*)(Qp + (size_t)row * HD_ + kh);
    aq[1] = *(const v16bf*)(Qp + (size_t)row * HD_ + 32 + kh);
  }

  v8f o[4];
  #pragma unroll
  for (int i = 0; i < 4; ++i) o[i] = (v8f){0.f,0.f,0.f,0.f,0.f,0.f,0.f,0.f};
  float m_i = -1e30f, l_i = 0.0f;    // maintained identically in lane pair (r, r+16)
  const float scale = 0.125f;

  int cur = 0;
  attn_stage(Kp, Vt, 0, &Ks[0][0], &Vst[0][0], t, wid);

  for (int kc = 0; kc < C_; kc += 64) {
    const bool more = (kc + 64) < C_;
    if (more) {
      attn_stage(Kp, Vt, kc + 64, &Ks[cur ^ 1][0], &Vst[cur ^ 1][0], t, wid);
      wait_tensor<2>();
      wait_async<8>();
    } else {
      wait_tensor<0>();
      wait_async<0>();
    }
    __syncthreads();

    // S = Q @ K^T
    #pragma unroll
    for (int nt = 0; nt < 4; ++nt) {
      const int key = nt * 16 + (lane & 15);
      const v16bf bk0 = *(const v16bf*)(&Ks[cur][key * 64 + kh]);
      const v16bf bk1 = *(const v16bf*)(&Ks[cur][key * 64 + 32 + kh]);
      v8f s = (v8f){0.f,0.f,0.f,0.f,0.f,0.f,0.f,0.f};
      s = wmma_bf16(aq[0], bk0, s);
      s = wmma_bf16(aq[1], bk1, s);
      #pragma unroll
      for (int e = 0; e < 8; ++e) {
        const int Mr = e + ((lane >> 4) ? 8 : 0);
        Sl[wid][Mr * 64 + nt * 16 + (lane & 15)] = s[e] * scale;
      }
    }
    __syncthreads();

    // online softmax: lane pair (r, r+16) shares row r, 32 columns each
    {
      const int r  = lane & 15;
      const int cb = (lane >> 4) * 32;
      const float* Srow = &Sl[wid][r * 64 + cb];
      float mx = -1e30f;
      for (int j = 0; j < 32; ++j) mx = fmaxf(mx, Srow[j]);
      mx = fmaxf(mx, __shfl_xor(mx, 16, 32));
      mx = fmaxf(mx, m_i);
      const float alpha = __expf(m_i - mx);
      float ls = 0.0f;
      unsigned short* Prow = &Pl[wid][r * 64 + cb];
      for (int j = 0; j < 32; ++j) {
        const float p = __expf(Srow[j] - mx);
        ls += p;
        Prow[j] = f2bf(p);
      }
      ls += __shfl_xor(ls, 16, 32);
      l_i = l_i * alpha + ls;
      m_i = mx;
      if (lane < 16) alpha_s[wid][r] = alpha;
    }
    __syncthreads();

    // rescale O, then O += P @ V
    #pragma unroll
    for (int nt = 0; nt < 4; ++nt)
      #pragma unroll
      for (int e = 0; e < 8; ++e)
        o[nt][e] *= alpha_s[wid][e + ((lane >> 4) ? 8 : 0)];

    const int prow = lane & 15;
    const v16bf ap0 = *(const v16bf*)(&Pl[wid][prow * 64 + kh]);
    const v16bf ap1 = *(const v16bf*)(&Pl[wid][prow * 64 + 32 + kh]);
    #pragma unroll
    for (int nt = 0; nt < 4; ++nt) {
      const int d = nt * 16 + (lane & 15);
      const v16bf bv0 = *(const v16bf*)(&Vst[cur][d * 64 + kh]);
      const v16bf bv1 = *(const v16bf*)(&Vst[cur][d * 64 + 32 + kh]);
      o[nt] = wmma_bf16(ap0, bv0, o[nt]);
      o[nt] = wmma_bf16(ap1, bv1, o[nt]);
    }
    __syncthreads();
    cur ^= 1;
  }

  if (lane < 16) linv_s[wid][lane] = 1.0f / l_i;
  __syncthreads();
  const int b = bh >> 4, h = bh & (H_ - 1);
  #pragma unroll
  for (int nt = 0; nt < 4; ++nt) {
    #pragma unroll
    for (int e = 0; e < 8; ++e) {
      const int Mr = e + ((lane >> 4) ? 8 : 0);
      const int c  = qbase + Mr;
      const float val = o[nt][e] * linv_s[wid][Mr];
      obf[((size_t)(b * C_ + c)) * D_ + h * HD_ + nt * 16 + (lane & 15)] = f2bf(val);
    }
  }
}

// ---------------------------------------------------------------------------
// Host pipeline (all on `stream`; ~109 MB of d_ws scratch).
// ---------------------------------------------------------------------------
extern "C" void kernel_launch(void* const* d_in, const int* in_sizes, int n_in,
                              void* d_out, int out_size, void* d_ws, size_t ws_size,
                              hipStream_t stream) {
  (void)in_sizes; (void)n_in; (void)out_size; (void)ws_size;
  const float* x      = (const float*)d_in[0];
  const float* ln_q_g = (const float*)d_in[1];
  const float* ln_q_b = (const float*)d_in[2];
  const float* Wq     = (const float*)d_in[3];
  const float* bq     = (const float*)d_in[4];
  const float* ln_kv_g= (const float*)d_in[5];
  const float* ln_kv_b= (const float*)d_in[6];
  const float* Wkv    = (const float*)d_in[7];
  const float* bkv    = (const float*)d_in[8];
  const float* Wout   = (const float*)d_in[9];
  const float* bout   = (const float*)d_in[10];
  const float* lnh_g  = (const float*)d_in[11];
  const float* lnh_b  = (const float*)d_in[12];
  const float* Wr     = (const float*)d_in[13];
  const float* br     = (const float*)d_in[14];
  float* out = (float*)d_out;

  char* w = (char*)d_ws;
  unsigned short* lnq   = (unsigned short*)(w + 0);          //  8 MB [BC,D]
  unsigned short* lnkv  = (unsigned short*)(w + 8388608);    //  8 MB
  unsigned short* wqt   = (unsigned short*)(w + 16777216);   //  2 MB [N,K]
  unsigned short* wkvt  = (unsigned short*)(w + 18874368);   //  4 MB [2D,K]
  unsigned short* woutt = (unsigned short*)(w + 23068672);   //  2 MB [N,K]
  float*          rad   = (float*)        (w + 25165824);    // 16 KB
  float*          qf    = (float*)        (w + 25182208);    // 16 MB [BC,D]
  float*          kvf   = (float*)        (w + 41959424);    // 32 MB [BC,2D]
  unsigned short* qn    = (unsigned short*)(w + 75513856);   //  8 MB [B,H,C,HD]
  unsigned short* kn    = (unsigned short*)(w + 83902464);   //  8 MB
  unsigned short* vbT   = (unsigned short*)(w + 92291072);   //  8 MB [B,H,HD,C]
  unsigned short* ob    = (unsigned short*)(w + 100679680);  //  8 MB [BC,D]

  k_prep<<<BC_, 256, 0, stream>>>(x, ln_q_g, ln_q_b, ln_kv_g, ln_kv_b, Wr, br,
                                  lnq, lnkv, rad);
  k_w2bf_t<<<(D_ * D_) / 256, 256, 0, stream>>>(Wq, wqt, D_, D_);
  k_w2bf_t<<<(D_ * 2 * D_) / 256, 256, 0, stream>>>(Wkv, wkvt, D_, 2 * D_);
  k_w2bf_t<<<(D_ * D_) / 256, 256, 0, stream>>>(Wout, woutt, D_, D_);

  k_gemm<<<dim3(D_ / 128, BC_ / 128), 256, 0, stream>>>(lnq, wqt, bq, qf,
                                                        BC_, D_, D_);
  k_gemm<<<dim3((2 * D_) / 128, BC_ / 128), 256, 0, stream>>>(lnkv, wkvt, bkv,
                                                              kvf, BC_, 2 * D_, D_);
  k_rope_ln<<<BC_ * H_, 32, 0, stream>>>(qf, kvf, rad, lnh_g, lnh_b, qn, kn, vbT);
  k_attn<<<(B_ * H_) * (C_ / 64), 128, 0, stream>>>(qn, kn, vbT, ob);
  k_gemm<<<dim3(D_ / 128, BC_ / 128), 256, 0, stream>>>(ob, woutt, bout, out,
                                                        BC_, D_, D_);
}
